// GemNetModel_53790170415615
// MI455X (gfx1250) — matrix operand order repaired and checked by reading, not compiled, not run on hardware
//
#include <hip/hip_runtime.h>

#define NN 50000
#define EE 800000
#define DDIM 128
#define HDIM 128
#define RDIM 64
#define LNUM 4
#define GNUM 64
#define CUTOFF_F 6.0f
#define GAMMA_F 10.0f

typedef __attribute__((ext_vector_type(16))) __bf16 v16bf;
typedef __attribute__((ext_vector_type(8)))  float  v8f;

union FragAB {
    v16bf v;
    uint4 q[2];
};
union FragC {
    v8f   v;
    float f[8];
};

// Native bf16 conversion -> v_cvt_pk_bf16_f32 on CDNA5.
__device__ __forceinline__ unsigned short f2bf(float f) {
    union { __bf16 h; unsigned short u; } b;
    b.h = (__bf16)f;
    return b.u;
}

__device__ __forceinline__ float silu_f(float v) {
    return v / (1.0f + __expf(-v));
}

// A-fragment (16x32 bf16, ISA layout): lane holds row (lane&15); first 8
// elements at K = kb..kb+7, second 8 at K = kb+16..kb+23 with kb = 8*(lane>=16).
__device__ __forceinline__ FragAB load_a_bf16(const unsigned short* src, int ka) {
    FragAB a;
    a.q[0] = *(const uint4*)(src + ka);
    a.q[1] = *(const uint4*)(src + ka + 16);
    return a;
}

// B-fragment (32x16 bf16): lane holds col (lane&15); first 8 elements at
// K = kb..kb+7, second 8 at K = kb+16..kb+23 with kb = 8*(lane>=16).
__device__ __forceinline__ FragAB load_b_bf16(const unsigned short* src) {
    FragAB b;
    b.q[0] = *(const uint4*)(src);
    b.q[1] = *(const uint4*)(src + 16);
    return b;
}

// Convert 16 gathered f32 (4 float4) into one bf16 A-fragment.
__device__ __forceinline__ FragAB cvt_a_f32(float4 f0, float4 f1, float4 f2, float4 f3) {
    FragAB a;
    a.v[0]  = (__bf16)f0.x; a.v[1]  = (__bf16)f0.y; a.v[2]  = (__bf16)f0.z; a.v[3]  = (__bf16)f0.w;
    a.v[4]  = (__bf16)f1.x; a.v[5]  = (__bf16)f1.y; a.v[6]  = (__bf16)f1.z; a.v[7]  = (__bf16)f1.w;
    a.v[8]  = (__bf16)f2.x; a.v[9]  = (__bf16)f2.y; a.v[10] = (__bf16)f2.z; a.v[11] = (__bf16)f2.w;
    a.v[12] = (__bf16)f3.x; a.v[13] = (__bf16)f3.y; a.v[14] = (__bf16)f3.z; a.v[15] = (__bf16)f3.w;
    return a;
}

#define WMMA_BF16(a, b, c) \
    __builtin_amdgcn_wmma_f32_16x16x32_bf16(false, (a).v, false, (b).v, (short)0, (c).v, false, false)

// ---------------------------------------------------------------------------
// Kernel 1: RBF expansion of edge distances -> bf16 [E, 64]
// ---------------------------------------------------------------------------
__global__ void rbf_kernel(const float* __restrict__ pos, const int* __restrict__ ei,
                           unsigned short* __restrict__ rbf) {
    int e = blockIdx.x * blockDim.x + threadIdx.x;
    if (e >= EE) return;
    int j = ei[e];        // row (source)
    int i = ei[EE + e];   // col (target)
    float dx = pos[j * 3 + 0] - pos[i * 3 + 0];
    float dy = pos[j * 3 + 1] - pos[i * 3 + 1];
    float dz = pos[j * 3 + 2] - pos[i * 3 + 2];
    float dist = sqrtf(dx * dx + dy * dy + dz * dz);
    const float step = CUTOFF_F / (float)(RDIM - 1);
    unsigned short* out = rbf + (size_t)e * RDIM;
#pragma unroll 8
    for (int k = 0; k < RDIM; ++k) {
        float d = dist - (float)k * step;
        out[k] = f2bf(__expf(-GAMMA_F * d * d));
    }
}

// ---------------------------------------------------------------------------
// Kernel 2 (per layer): fold W_rbf into W_e1's e-block, transpose all edge /
// update weights into bf16 images laid out for B-fragment loads.
// ---------------------------------------------------------------------------
__global__ void fold_kernel(const float* __restrict__ We1, const float* __restrict__ Wrbf,
                            const float* __restrict__ brbf, const float* __restrict__ be1,
                            const float* __restrict__ We2, const float* __restrict__ Wu1,
                            const float* __restrict__ Wu2,
                            unsigned short* __restrict__ wcatT, unsigned short* __restrict__ w2T,
                            unsigned short* __restrict__ wu1T, unsigned short* __restrict__ wu2T,
                            float* __restrict__ bias_cat) {
    int tid = blockIdx.x * blockDim.x + threadIdx.x;
    if (tid < 320 * 128) {
        int n = tid / 320, k = tid % 320;
        float v;
        if (k < 256) {
            v = We1[k * 128 + n];
        } else {
            int r = k - 256;
            float s = 0.0f;
#pragma clang loop unroll(disable)
            for (int j = 0; j < 128; ++j) s += Wrbf[r * 128 + j] * We1[(256 + j) * 128 + n];
            v = s;
        }
        wcatT[n * 320 + k] = f2bf(v);
        return;
    }
    tid -= 320 * 128;
    if (tid < 128 * 128) { int n = tid >> 7, k = tid & 127; w2T[n * 128 + k] = f2bf(We2[k * 128 + n]); return; }
    tid -= 128 * 128;
    if (tid < 128 * 128) { int n = tid >> 7, k = tid & 127; wu1T[n * 128 + k] = f2bf(Wu1[k * 128 + n]); return; }
    tid -= 128 * 128;
    if (tid < 128 * 128) { int n = tid >> 7, k = tid & 127; wu2T[n * 128 + k] = f2bf(Wu2[k * 128 + n]); return; }
    tid -= 128 * 128;
    if (tid < 128) {
        float s = be1[tid];
#pragma clang loop unroll(disable)
        for (int j = 0; j < 128; ++j) s += brbf[j] * We1[(256 + j) * 128 + tid];
        bias_cat[tid] = s;
    }
}

// ---------------------------------------------------------------------------
// Kernel 3: h = bf16(x @ W_node + b_node).  8 waves/block, one 16-row M-tile
// per wave, W_node transposed into LDS as bf16 (coalesced global reads).
// All A data (the lane's 64 floats) loaded in one burst, then converted.
// ---------------------------------------------------------------------------
__global__ void __launch_bounds__(256) node_gemm_kernel(
    const float* __restrict__ x, const float* __restrict__ Wn,
    const float* __restrict__ bn, unsigned short* __restrict__ hout) {
    __shared__ unsigned short wT[128 * 128];   // 32 KB
    for (int idx = threadIdx.x; idx < 128 * 128; idx += 256) {
        int k = idx >> 7, n = idx & 127;       // coalesced read of Wn[idx]
        wT[n * 128 + k] = f2bf(Wn[idx]);
    }
    __syncthreads();

    int wave = threadIdx.x >> 5;
    int lane = threadIdx.x & 31;
    int li = lane & 15, hi = lane >> 4;
    int mbase = blockIdx.x * 128 + wave * 16;
    int row = mbase + li;
    int rowc = row < NN ? row : NN - 1;

    const float* arow = x + (size_t)rowc * 128;

    // burst-load all 16 float4s for the 4 A-fragments, then convert
    float4 fa[16];
#pragma unroll
    for (int ks = 0; ks < 4; ++ks) {
        int ka = ks * 32 + hi * 8;
        fa[4 * ks + 0] = *(const float4*)(arow + ka);
        fa[4 * ks + 1] = *(const float4*)(arow + ka + 4);
        fa[4 * ks + 2] = *(const float4*)(arow + ka + 16);
        fa[4 * ks + 3] = *(const float4*)(arow + ka + 20);
    }
    FragAB afr[4];
#pragma unroll
    for (int ks = 0; ks < 4; ++ks)
        afr[ks] = cvt_a_f32(fa[4 * ks], fa[4 * ks + 1], fa[4 * ks + 2], fa[4 * ks + 3]);

    FragC c[8];
#pragma unroll
    for (int nt = 0; nt < 8; ++nt)
#pragma unroll
        for (int r = 0; r < 8; ++r) c[nt].f[r] = 0.0f;

#pragma unroll
    for (int ks = 0; ks < 4; ++ks) {
        const unsigned short* bbase = &wT[li * 128 + ks * 32 + hi * 8];
#pragma unroll
        for (int nt = 0; nt < 8; ++nt) {
            FragAB b = load_b_bf16(bbase + nt * 16 * 128);   // const offset -> imm
            c[nt].v = WMMA_BF16(afr[ks], b, c[nt]);
        }
    }

    float biasv[8];
#pragma unroll
    for (int nt = 0; nt < 8; ++nt) biasv[nt] = bn[nt * 16 + li];
#pragma unroll
    for (int r = 0; r < 8; ++r) {
        int ro = mbase + r + 8 * hi;
        if (ro < NN) {
            unsigned short* p = hout + (size_t)ro * 128 + li;
#pragma unroll
            for (int nt = 0; nt < 8; ++nt) p[nt * 16] = f2bf(c[nt].f[r] + biasv[nt]);
        }
    }
}

// ---------------------------------------------------------------------------
// Kernel 4: generic f32 zero
// ---------------------------------------------------------------------------
__global__ void zero_kernel(float* __restrict__ p, int n) {
    int i = blockIdx.x * blockDim.x + threadIdx.x;
    if (i < n) p[i] = 0.0f;
}

// ---------------------------------------------------------------------------
// Kernel 5: the edge message MLP.  One wave == one 16-edge tile.
//   msg = [h[col] | h[row] | rbf]  (K = 320, W_rbf folded in)
//   m   = silu(msg @ Wcat + bias_cat) @ W_e2 + b_e2
//   agg[col] += m   (f32 atomics)
// All 10 A-fragments gathered up front (one load burst), then 80 WMMAs.
// ---------------------------------------------------------------------------
__global__ void __launch_bounds__(128) message_kernel(
    const unsigned short* __restrict__ hb,     // [N,128] bf16
    const unsigned short* __restrict__ rbf,    // [E,64]  bf16
    const int* __restrict__ ei,                // [2,E]
    const unsigned short* __restrict__ wcatT,  // [128][320] bf16
    const float* __restrict__ bias_cat,        // [128]
    const unsigned short* __restrict__ w2T,    // [128][128] bf16
    const float* __restrict__ be2,             // [128]
    float* __restrict__ agg)                   // [N,128] f32
{
    __shared__ unsigned short m1s[4][16 * 128];   // 4 waves x 4 KB

    int wave = threadIdx.x >> 5;
    int lane = threadIdx.x & 31;
    int li = lane & 15, hi = lane >> 4;
    int tile = blockIdx.x * 4 + wave;
    int myE = tile * 16 + li;

    int ri = ei[myE];        // source j
    int ci = ei[EE + myE];   // target i

    const unsigned short* acol = hb + (size_t)ci * 128;
    const unsigned short* arow = hb + (size_t)ri * 128;
    const unsigned short* arbf = rbf + (size_t)myE * 64;
    __builtin_prefetch(acol, 0, 3);
    __builtin_prefetch(arow, 0, 3);
    __builtin_prefetch(arbf, 0, 3);

    // burst-gather all A-fragments for the K=320 loop
    FragAB afr[10];
#pragma unroll
    for (int ks = 0; ks < 4; ++ks) afr[ks] = load_a_bf16(acol, ks * 32 + hi * 8);
#pragma unroll
    for (int ks = 0; ks < 4; ++ks) afr[4 + ks] = load_a_bf16(arow, ks * 32 + hi * 8);
#pragma unroll
    for (int ks = 0; ks < 2; ++ks) afr[8 + ks] = load_a_bf16(arbf, ks * 32 + hi * 8);

    FragC c1[8];
#pragma unroll
    for (int nt = 0; nt < 8; ++nt)
#pragma unroll
        for (int r = 0; r < 8; ++r) c1[nt].f[r] = 0.0f;

#pragma unroll
    for (int ks = 0; ks < 10; ++ks) {
        const unsigned short* bbase = wcatT + li * 320 + ks * 32 + hi * 8;
#pragma unroll
        for (int nt = 0; nt < 8; ++nt) {
            FragAB b = load_b_bf16(bbase + nt * 16 * 320);   // const offset -> imm
            c1[nt].v = WMMA_BF16(afr[ks], b, c1[nt]);
        }
    }

    // bias + silu, transpose through LDS to A-layout for GEMM2
    float biasv[8];
#pragma unroll
    for (int nt = 0; nt < 8; ++nt) biasv[nt] = bias_cat[nt * 16 + li];
#pragma unroll
    for (int r = 0; r < 8; ++r) {
        unsigned short* p = &m1s[wave][(r + 8 * hi) * 128 + li];
#pragma unroll
        for (int nt = 0; nt < 8; ++nt) p[nt * 16] = f2bf(silu_f(c1[nt].f[r] + biasv[nt]));
    }
    __syncthreads();

    const unsigned short* am = &m1s[wave][li * 128];
    FragAB a2[4];
#pragma unroll
    for (int ks = 0; ks < 4; ++ks) a2[ks] = load_a_bf16(am, ks * 32 + hi * 8);

    FragC c2[8];
#pragma unroll
    for (int nt = 0; nt < 8; ++nt)
#pragma unroll
        for (int r = 0; r < 8; ++r) c2[nt].f[r] = 0.0f;

#pragma unroll
    for (int ks = 0; ks < 4; ++ks) {
        const unsigned short* bbase = w2T + li * 128 + ks * 32 + hi * 8;
#pragma unroll
        for (int nt = 0; nt < 8; ++nt) {
            FragAB b = load_b_bf16(bbase + nt * 16 * 128);
            c2[nt].v = WMMA_BF16(a2[ks], b, c2[nt]);
        }
    }

    // broadcast target index of each output row, then scatter-add
    int cm[8];
#pragma unroll
    for (int r = 0; r < 8; ++r) cm[r] = __shfl(ci, r + 8 * hi, 32);

#pragma unroll
    for (int nt = 0; nt < 8; ++nt) biasv[nt] = be2[nt * 16 + li];
#pragma unroll
    for (int r = 0; r < 8; ++r) {
        float* basep = agg + (size_t)cm[r] * 128 + li;
#pragma unroll
        for (int nt = 0; nt < 8; ++nt) {
            atomicAdd(basep + nt * 16, c2[nt].f[r] + biasv[nt]);   // const offset -> imm
        }
    }
}

// ---------------------------------------------------------------------------
// Kernel 6: x_new = silu(agg @ W_u1 + b_u1) @ W_u2 + b_u2   (fused two GEMMs)
// ---------------------------------------------------------------------------
__global__ void __launch_bounds__(256) update_kernel(
    const float* __restrict__ agg, const unsigned short* __restrict__ wu1T,
    const float* __restrict__ bu1, const unsigned short* __restrict__ wu2T,
    const float* __restrict__ bu2, float* __restrict__ xout) {
    __shared__ unsigned short m1s[8][16 * 128];   // 8 waves x 4 KB

    int wave = threadIdx.x >> 5;
    int lane = threadIdx.x & 31;
    int li = lane & 15, hi = lane >> 4;
    int mbase = blockIdx.x * 128 + wave * 16;
    int row = mbase + li;
    int rowc = row < NN ? row : NN - 1;

    const float* arow = agg + (size_t)rowc * 128;

    float4 fa[16];
#pragma unroll
    for (int ks = 0; ks < 4; ++ks) {
        int ka = ks * 32 + hi * 8;
        fa[4 * ks + 0] = *(const float4*)(arow + ka);
        fa[4 * ks + 1] = *(const float4*)(arow + ka + 4);
        fa[4 * ks + 2] = *(const float4*)(arow + ka + 16);
        fa[4 * ks + 3] = *(const float4*)(arow + ka + 20);
    }
    FragAB afr[4];
#pragma unroll
    for (int ks = 0; ks < 4; ++ks)
        afr[ks] = cvt_a_f32(fa[4 * ks], fa[4 * ks + 1], fa[4 * ks + 2], fa[4 * ks + 3]);

    FragC c1[8];
#pragma unroll
    for (int nt = 0; nt < 8; ++nt)
#pragma unroll
        for (int r = 0; r < 8; ++r) c1[nt].f[r] = 0.0f;

#pragma unroll
    for (int ks = 0; ks < 4; ++ks) {
        const unsigned short* bbase = wu1T + li * 128 + ks * 32 + hi * 8;
#pragma unroll
        for (int nt = 0; nt < 8; ++nt) {
            FragAB b = load_b_bf16(bbase + nt * 16 * 128);
            c1[nt].v = WMMA_BF16(afr[ks], b, c1[nt]);
        }
    }

    float biasv[8];
#pragma unroll
    for (int nt = 0; nt < 8; ++nt) biasv[nt] = bu1[nt * 16 + li];
#pragma unroll
    for (int r = 0; r < 8; ++r) {
        unsigned short* p = &m1s[wave][(r + 8 * hi) * 128 + li];
#pragma unroll
        for (int nt = 0; nt < 8; ++nt) p[nt * 16] = f2bf(silu_f(c1[nt].f[r] + biasv[nt]));
    }
    __syncthreads();

    const unsigned short* am = &m1s[wave][li * 128];
    FragAB a2[4];
#pragma unroll
    for (int ks = 0; ks < 4; ++ks) a2[ks] = load_a_bf16(am, ks * 32 + hi * 8);

    FragC c2[8];
#pragma unroll
    for (int nt = 0; nt < 8; ++nt)
#pragma unroll
        for (int r = 0; r < 8; ++r) c2[nt].f[r] = 0.0f;

#pragma unroll
    for (int ks = 0; ks < 4; ++ks) {
        const unsigned short* bbase = wu2T + li * 128 + ks * 32 + hi * 8;
#pragma unroll
        for (int nt = 0; nt < 8; ++nt) {
            FragAB b = load_b_bf16(bbase + nt * 16 * 128);
            c2[nt].v = WMMA_BF16(a2[ks], b, c2[nt]);
        }
    }

#pragma unroll
    for (int nt = 0; nt < 8; ++nt) biasv[nt] = bu2[nt * 16 + li];
#pragma unroll
    for (int r = 0; r < 8; ++r) {
        int ro = mbase + r + 8 * hi;
        if (ro < NN) {
            float* p = xout + (size_t)ro * 128 + li;
#pragma unroll
            for (int nt = 0; nt < 8; ++nt) p[nt * 16] = c2[nt].f[r] + biasv[nt];
        }
    }
}

// ---------------------------------------------------------------------------
// Kernel 7: graph pooling  g[batch[n]] += x[n]
// ---------------------------------------------------------------------------
__global__ void pool_kernel(const float* __restrict__ xf, const int* __restrict__ batch,
                            float* __restrict__ g) {
    int idx = blockIdx.x * blockDim.x + threadIdx.x;
    if (idx >= NN * 128) return;
    int n = idx >> 7, d = idx & 127;
    atomicAdd(g + batch[n] * 128 + d, xf[idx]);
}

// ---------------------------------------------------------------------------
// Kernel 8: readout  out = silu(g @ W_f1 + b_f1) @ W_f2 + b_f2   (tiny)
// ---------------------------------------------------------------------------
__global__ void readout_kernel(const float* __restrict__ g, const float* __restrict__ Wf1,
                               const float* __restrict__ bf1, const float* __restrict__ Wf2,
                               const float* __restrict__ bf2, float* __restrict__ out) {
    int gi = blockIdx.x * blockDim.x + threadIdx.x;
    if (gi >= GNUM) return;
    float acc = bf2[0];
#pragma clang loop unroll(disable)
    for (int h = 0; h < 128; ++h) {
        float s = bf1[h];
#pragma clang loop unroll(disable)
        for (int d = 0; d < 128; ++d) s += g[gi * 128 + d] * Wf1[d * 128 + h];
        acc += silu_f(s) * Wf2[h];
    }
    out[gi] = acc;
}

// ---------------------------------------------------------------------------
extern "C" void kernel_launch(void* const* d_in, const int* in_sizes, int n_in,
                              void* d_out, int out_size, void* d_ws, size_t ws_size,
                              hipStream_t stream) {
    (void)in_sizes; (void)n_in; (void)out_size; (void)ws_size;

    const float* x_in   = (const float*)d_in[0];
    const float* pos    = (const float*)d_in[1];
    const int*   ei     = (const int*)d_in[2];
    const int*   batch  = (const int*)d_in[3];
    const float* W_node = (const float*)d_in[4];
    const float* b_node = (const float*)d_in[5];
    const float* W_rbf  = (const float*)d_in[6];
    const float* b_rbf  = (const float*)d_in[7];
    const float* W_e1   = (const float*)d_in[8];
    const float* b_e1   = (const float*)d_in[9];
    const float* W_e2   = (const float*)d_in[10];
    const float* b_e2   = (const float*)d_in[11];
    const float* W_u1   = (const float*)d_in[12];
    const float* b_u1   = (const float*)d_in[13];
    const float* W_u2   = (const float*)d_in[14];
    const float* b_u2   = (const float*)d_in[15];
    const float* W_f1   = (const float*)d_in[16];
    const float* b_f1   = (const float*)d_in[17];
    const float* W_f2   = (const float*)d_in[18];
    const float* b_f2   = (const float*)d_in[19];

    char* w = (char*)d_ws;
    auto carve = [&](size_t bytes) -> char* {
        char* p = w;
        w += (bytes + 255) & ~(size_t)255;
        return p;
    };
    unsigned short* rbf   = (unsigned short*)carve((size_t)EE * RDIM * 2);   // 102.4 MB
    unsigned short* hb    = (unsigned short*)carve((size_t)NN * HDIM * 2);   // 12.8 MB
    float* agg            = (float*)carve((size_t)NN * HDIM * 4);            // 25.6 MB
    float* xb0            = (float*)carve((size_t)NN * DDIM * 4);            // 25.6 MB
    float* xb1            = (float*)carve((size_t)NN * DDIM * 4);            // 25.6 MB
    unsigned short* wcatT = (unsigned short*)carve((size_t)320 * 128 * 2);
    unsigned short* w2T   = (unsigned short*)carve((size_t)128 * 128 * 2);
    unsigned short* wu1T  = (unsigned short*)carve((size_t)128 * 128 * 2);
    unsigned short* wu2T  = (unsigned short*)carve((size_t)128 * 128 * 2);
    float* bias_cat       = (float*)carve(128 * 4);
    float* gbuf           = (float*)carve((size_t)GNUM * DDIM * 4);

    rbf_kernel<<<(EE + 255) / 256, 256, 0, stream>>>(pos, ei, rbf);

    const int foldN = 320 * 128 + 3 * 128 * 128 + 128;
    const float* xcur = x_in;
    float* xnext = xb0;

    for (int l = 0; l < LNUM; ++l) {
        const float* We1l  = W_e1 + (size_t)l * 384 * 128;
        const float* Wrbfl = W_rbf + (size_t)l * 64 * 128;
        const float* brbfl = b_rbf + (size_t)l * 128;
        const float* be1l  = b_e1 + (size_t)l * 128;
        const float* We2l  = W_e2 + (size_t)l * 128 * 128;
        const float* Wu1l  = W_u1 + (size_t)l * 128 * 128;
        const float* Wu2l  = W_u2 + (size_t)l * 128 * 128;

        fold_kernel<<<(foldN + 255) / 256, 256, 0, stream>>>(
            We1l, Wrbfl, brbfl, be1l, We2l, Wu1l, Wu2l,
            wcatT, w2T, wu1T, wu2T, bias_cat);

        node_gemm_kernel<<<(NN + 127) / 128, 256, 0, stream>>>(
            xcur, W_node + (size_t)l * 128 * 128, b_node + (size_t)l * 128, hb);

        zero_kernel<<<(NN * HDIM + 255) / 256, 256, 0, stream>>>(agg, NN * HDIM);

        message_kernel<<<EE / 64, 128, 0, stream>>>(
            hb, rbf, ei, wcatT, bias_cat, w2T, b_e2 + (size_t)l * 128, agg);

        update_kernel<<<(NN + 127) / 128, 256, 0, stream>>>(
            agg, wu1T, b_u1 + (size_t)l * 128, wu2T, b_u2 + (size_t)l * 128, xnext);

        xcur = xnext;
        xnext = (xcur == xb0) ? xb1 : xb0;
    }

    zero_kernel<<<(GNUM * DDIM + 255) / 256, 256, 0, stream>>>(gbuf, GNUM * DDIM);
    pool_kernel<<<(NN * DDIM + 255) / 256, 256, 0, stream>>>(xcur, batch, gbuf);
    readout_kernel<<<1, 64, 0, stream>>>(gbuf, W_f1, b_f1, W_f2, b_f2, (float*)d_out);
}
